// SPGNN_56684978372727
// MI455X (gfx1250) — matrix operand order repaired and checked by reading, not compiled, not run on hardware
//
#include <hip/hip_runtime.h>

typedef __attribute__((ext_vector_type(16))) _Float16 v16h;
typedef __attribute__((ext_vector_type(8)))  _Float16 v8h;
typedef __attribute__((ext_vector_type(4)))  _Float16 v4h;
typedef __attribute__((ext_vector_type(8)))  float    v8f;
typedef __attribute__((ext_vector_type(4)))  float    v4f;

#define HDIM 128
#define BN_EPS 1e-5f
#define MT 4                // m-tiles (of 16 rows) per block -> 64 rows/block

__device__ __forceinline__ v8f zero8() {
    v8f c;
#pragma unroll
    for (int i = 0; i < 8; ++i) c[i] = 0.0f;
    return c;
}

__device__ __forceinline__ v8f wmma16(v16h a, v16h b, v8f c) {
    return __builtin_amdgcn_wmma_f32_16x16x32_f16(
        /*neg_a=*/false, a, /*neg_b=*/false, b,
        /*c_mod=*/(short)0, c, /*reuse_a=*/false, /*reuse_b=*/false);
}

// A fragment (16x32 f16) from f32 row-major [*, 128].
// lane 0-15: row=lane,    K = kc*32 + {0..7, 16..23}
// lane 16-31: row=lane-16, K = kc*32 + {8..15, 24..31}
__device__ __forceinline__ v16h load_a_f32(const float* A, int row, int kc, int lane) {
    const float* p = A + (size_t)row * HDIM + kc * 32 + ((lane >> 4) << 3);
    v4f x0 = *(const v4f*)(p);
    v4f x1 = *(const v4f*)(p + 4);
    v4f x2 = *(const v4f*)(p + 16);
    v4f x3 = *(const v4f*)(p + 20);
    v16h a;
#pragma unroll
    for (int i = 0; i < 4; ++i) {
        a[i]      = (_Float16)x0[i];
        a[4 + i]  = (_Float16)x1[i];
        a[8 + i]  = (_Float16)x2[i];
        a[12 + i] = (_Float16)x3[i];
    }
    return a;
}

// A fragment from f16 row-major LDS tile [16][128]
__device__ __forceinline__ v16h load_a_lds(const _Float16* U, int row, int kc, int lane) {
    const _Float16* p = U + row * HDIM + kc * 32 + ((lane >> 4) << 3);
    v8h x0 = *(const v8h*)(p);
    v8h x1 = *(const v8h*)(p + 16);
    v16h a;
#pragma unroll
    for (int i = 0; i < 8; ++i) { a[i] = x0[i]; a[8 + i] = x1[i]; }
    return a;
}

// B fragment from pre-packed buffer: tile (kc, nt), 512 halfs per tile, lane-major 16 halfs.
__device__ __forceinline__ v16h load_b_pk(const _Float16* P, int kc, int nt, int lane) {
    return *(const v16h*)(P + (((kc * 8 + nt) * 32 + lane) << 4));
}

// ---- pack f32 [128,128] row-major -> WMMA B fragment order (f16) ----
__global__ void pack_w_kernel(const float* __restrict__ W, _Float16* __restrict__ P) {
    int mat  = blockIdx.y;
    int d    = blockIdx.x * blockDim.x + threadIdx.x;   // 0..16383
    int tile = d >> 9;
    int rem  = d & 511;
    int lane = rem >> 4;
    int i    = rem & 15;
    int kt = tile >> 3, nt = tile & 7;
    int k = kt * 32 + ((lane >> 4) << 3) + (i & 7) + ((i >> 3) << 4);   // lanes 16-31 hold K=8..15,24..31
    int n = nt * 16 + (lane & 15);
    P[(size_t)mat * 16384 + d] = (_Float16)W[(size_t)mat * 16384 + k * HDIM + n];
}

// ---- embedding: h16 = f16(x @ embW + emb_b), 64 rows/block, B-frag reuse ----
__global__ __launch_bounds__(256) void emb_gemm_kernel(
    const float* __restrict__ X, const _Float16* __restrict__ Pw,
    const float* __restrict__ bias, _Float16* __restrict__ Hout, int N) {
    int lane = threadIdx.x & 31, wave = threadIdx.x >> 5;
    int m0 = blockIdx.x * (16 * MT);
    int rowc[MT];
#pragma unroll
    for (int mt = 0; mt < MT; ++mt) {
        int r = m0 + mt * 16 + (lane & 15);
        rowc[mt] = r < N ? r : N - 1;
    }
    v8f acc[MT];
#pragma unroll
    for (int mt = 0; mt < MT; ++mt) acc[mt] = zero8();
#pragma unroll
    for (int kc = 0; kc < 4; ++kc) {
        v16h b = load_b_pk(Pw, kc, wave, lane);
#pragma unroll
        for (int mt = 0; mt < MT; ++mt)
            acc[mt] = wmma16(load_a_f32(X, rowc[mt], kc, lane), b, acc[mt]);
    }
    int col = wave * 16 + (lane & 15);
    float bv = bias[col];
#pragma unroll
    for (int mt = 0; mt < MT; ++mt) {
        int rb = m0 + mt * 16 + ((lane >> 4) << 3);
#pragma unroll
        for (int v = 0; v < 8; ++v) {
            int r = rb + v;
            if (r < N) Hout[(size_t)r * HDIM + col] = (_Float16)(acc[mt][v] + bv);
        }
    }
}

// ---- edge scatter: agg3[order][dst] += h16[src]  (1 wave per edge, 4 cols/lane) ----
__global__ void scatter_kernel(const int* __restrict__ src, const int* __restrict__ dst,
                               const int* __restrict__ ord, const _Float16* __restrict__ h16,
                               float* __restrict__ agg3, int E, int N) {
    int gid = blockIdx.x * blockDim.x + threadIdx.x;
    int e = gid >> 5;
    if (e >= E) return;
    int lane = gid & 31;
    int s = src[e], d = dst[e], k = ord[e];
    v4h hv = *(const v4h*)(h16 + (size_t)s * HDIM + lane * 4);
    float* ap = agg3 + ((size_t)k * N + d) * HDIM + lane * 4;
#pragma unroll
    for (int c = 0; c < 4; ++c) atomicAdd(ap + c, (float)hv[c]);
}

// ---- fused layer: v = relu( (Σ_k agg_k @ convW_k) @ mlpW + mlp_b ), + BN partial sums ----
__global__ __launch_bounds__(256) void layer_fused_kernel(
    const float* __restrict__ agg3, const _Float16* __restrict__ Pconv,
    const _Float16* __restrict__ Pmlp, const float* __restrict__ mbias,
    float* __restrict__ vbuf, float* __restrict__ colsum, float* __restrict__ colsq,
    int N, int K1) {
    __shared__ _Float16 ush[16 * MT * HDIM];            // 64x128 f16 = 16 KB
    int lane = threadIdx.x & 31, wave = threadIdx.x >> 5;
    int m0 = blockIdx.x * (16 * MT);
    int rowc[MT];
#pragma unroll
    for (int mt = 0; mt < MT; ++mt) {
        int r = m0 + mt * 16 + (lane & 15);
        rowc[mt] = r < N ? r : N - 1;
    }

    // GEMM1: u = Σ_k agg_k @ W_k   (B fragment loaded once, reused across MT)
    v8f acc[MT];
#pragma unroll
    for (int mt = 0; mt < MT; ++mt) acc[mt] = zero8();
    for (int k = 0; k < K1; ++k) {
        const float* A = agg3 + (size_t)k * N * HDIM;
        const _Float16* P = Pconv + (size_t)k * 16384;
#pragma unroll
        for (int kc = 0; kc < 4; ++kc) {
            v16h b = load_b_pk(P, kc, wave, lane);
#pragma unroll
            for (int mt = 0; mt < MT; ++mt)
                acc[mt] = wmma16(load_a_f32(A, rowc[mt], kc, lane), b, acc[mt]);
        }
    }
    // stage u as f16 in LDS (row-major [64][128])
    int col = lane & 15;
    int rb = (lane >> 4) << 3;
#pragma unroll
    for (int mt = 0; mt < MT; ++mt)
#pragma unroll
        for (int v = 0; v < 8; ++v)
            ush[(mt * 16 + rb + v) * HDIM + wave * 16 + col] = (_Float16)acc[mt][v];
    __syncthreads();

    // GEMM2: v = u @ mlpW
    v8f acc2[MT];
#pragma unroll
    for (int mt = 0; mt < MT; ++mt) acc2[mt] = zero8();
#pragma unroll
    for (int kc = 0; kc < 4; ++kc) {
        v16h b = load_b_pk(Pmlp, kc, wave, lane);
#pragma unroll
        for (int mt = 0; mt < MT; ++mt)
            acc2[mt] = wmma16(load_a_lds(ush + mt * 16 * HDIM, lane & 15, kc, lane), b, acc2[mt]);
    }

    int gcol = wave * 16 + col;
    float bv = mbias[gcol];
    float s1 = 0.0f, s2 = 0.0f;
#pragma unroll
    for (int mt = 0; mt < MT; ++mt) {
#pragma unroll
        for (int v = 0; v < 8; ++v) {
            int r = m0 + mt * 16 + rb + v;
            float val = acc2[mt][v] + bv;
            val = val > 0.0f ? val : 0.0f;
            if (r < N) {
                vbuf[(size_t)r * HDIM + gcol] = val;
                s1 += val;
                s2 += val * val;
            }
        }
    }
    atomicAdd(colsum + gcol, s1);
    atomicAdd(colsq + gcol, s2);
}

// ---- BN: scale/shift from batch stats ----
__global__ void bn_finalize_kernel(const float* __restrict__ colsum, const float* __restrict__ colsq,
                                   const float* __restrict__ g, const float* __restrict__ b,
                                   float* __restrict__ ss, int N) {
    int c = threadIdx.x;
    float inv = 1.0f / (float)N;
    float mu  = colsum[c] * inv;
    float var = colsq[c] * inv - mu * mu;
    float sc  = g[c] * rsqrtf(var + BN_EPS);
    ss[c] = sc;
    ss[HDIM + c] = b[c] - mu * sc;
}

__global__ void bn_apply_kernel(const float* __restrict__ vbuf, const float* __restrict__ ss,
                                _Float16* __restrict__ h16, int total) {
    int i = blockIdx.x * blockDim.x + threadIdx.x;
    if (i >= total) return;
    int col = i & (HDIM - 1);
    h16[i] = (_Float16)(vbuf[i] * ss[col] + ss[HDIM + col]);
}

// ---- graph pooling: pooled[graph_id[n]] += h[n] ----
__global__ void pool_kernel(const int* __restrict__ gidx, const _Float16* __restrict__ h16,
                            float* __restrict__ pooled, int N) {
    int gid = blockIdx.x * blockDim.x + threadIdx.x;
    int n = gid >> 5;
    if (n >= N) return;
    int lane = gid & 31;
    int g = gidx[n];
    v4h hv = *(const v4h*)(h16 + (size_t)n * HDIM + lane * 4);
    float* pp = pooled + (size_t)g * HDIM + lane * 4;
#pragma unroll
    for (int c = 0; c < 4; ++c) atomicAdd(pp + c, (float)hv[c]);
}

// ---- classifier: out = relu(pooled@W1+b1)@W2+b2 (tiny, scalar) ----
__global__ void classifier_kernel(const float* __restrict__ pooled,
                                  const float* __restrict__ W1, const float* __restrict__ b1,
                                  const float* __restrict__ W2, const float* __restrict__ b2,
                                  float* __restrict__ out, int Bg, int C, int Hh) {
    extern __shared__ float zsh[];
    int total1 = Bg * Hh;
    for (int t = threadIdx.x; t < total1; t += blockDim.x) {
        int b = t / Hh, j = t - b * Hh;
        float acc = b1[j];
        const float* pr = pooled + (size_t)b * HDIM;
        for (int i = 0; i < HDIM; ++i) acc += pr[i] * W1[i * Hh + j];
        zsh[t] = acc > 0.0f ? acc : 0.0f;
    }
    __syncthreads();
    int total2 = Bg * C;
    for (int t = threadIdx.x; t < total2; t += blockDim.x) {
        int b = t / C, cc = t - b * C;
        float acc = b2[cc];
        const float* zr = zsh + b * Hh;
        for (int j = 0; j < Hh; ++j) acc += zr[j] * W2[j * C + cc];
        out[t] = acc;
    }
}

extern "C" void kernel_launch(void* const* d_in, const int* in_sizes, int n_in,
                              void* d_out, int out_size, void* d_ws, size_t ws_size,
                              hipStream_t stream) {
    (void)n_in; (void)ws_size;
    const int*   src     = (const int*)d_in[0];
    const int*   dst     = (const int*)d_in[1];
    const int*   order   = (const int*)d_in[2];
    const int*   graphid = (const int*)d_in[3];
    const float* x       = (const float*)d_in[4];
    const float* emb_W   = (const float*)d_in[5];
    const float* emb_b   = (const float*)d_in[6];
    const float* conv_W  = (const float*)d_in[7];
    const float* mlp_W   = (const float*)d_in[8];
    const float* mlp_b   = (const float*)d_in[9];
    const float* bn_g    = (const float*)d_in[10];
    const float* bn_b    = (const float*)d_in[11];
    const float* cls_W1  = (const float*)d_in[12];
    const float* cls_b1  = (const float*)d_in[13];
    const float* cls_W2  = (const float*)d_in[14];
    const float* cls_b2  = (const float*)d_in[15];
    float* out = (float*)d_out;

    const int E  = in_sizes[0];
    const int N  = in_sizes[3];
    const int L  = in_sizes[8] / (HDIM * HDIM);
    const int K1 = in_sizes[7] / (L * HDIM * HDIM);     // K+1 = 3
    const int Hh = in_sizes[13];                        // H/2 = 64
    const int C  = in_sizes[15];                        // 10
    const int Bg = out_size / C;                        // 64

    // ---- carve workspace ----
    unsigned char* wsp = (unsigned char*)d_ws;
    size_t off = 0;
    auto carve = [&](size_t bytes) -> void* {
        void* p = wsp + off;
        off = (off + bytes + 255) & ~(size_t)255;
        return p;
    };
    _Float16* packEmb  = (_Float16*)carve((size_t)HDIM * HDIM * 2);
    _Float16* packConv = (_Float16*)carve((size_t)L * K1 * HDIM * HDIM * 2);
    _Float16* packMlp  = (_Float16*)carve((size_t)L * HDIM * HDIM * 2);
    _Float16* h16      = (_Float16*)carve((size_t)N * HDIM * 2);
    float*    agg3     = (float*)carve((size_t)K1 * N * HDIM * 4);
    float*    vbuf     = (float*)carve((size_t)N * HDIM * 4);
    float*    colsum   = (float*)carve(HDIM * 4);
    float*    colsq    = (float*)carve(HDIM * 4);
    float*    ss       = (float*)carve(2 * HDIM * 4);
    float*    pooled   = (float*)carve((size_t)Bg * HDIM * 4);

    // ---- pack weights into WMMA B-fragment order ----
    pack_w_kernel<<<dim3(64, 1),       256, 0, stream>>>(emb_W,  packEmb);
    pack_w_kernel<<<dim3(64, L * K1),  256, 0, stream>>>(conv_W, packConv);
    pack_w_kernel<<<dim3(64, L),       256, 0, stream>>>(mlp_W,  packMlp);

    const int mblocks = (N + 16 * MT - 1) / (16 * MT);

    // ---- embedding ----
    emb_gemm_kernel<<<mblocks, 256, 0, stream>>>(x, packEmb, emb_b, h16, N);

    // ---- layers ----
    const int scatterBlocks = (int)(((size_t)E * 32 + 255) / 256);
    for (int l = 0; l < L; ++l) {
        hipMemsetAsync(agg3,   0, (size_t)K1 * N * HDIM * 4, stream);
        hipMemsetAsync(colsum, 0, HDIM * 4, stream);
        hipMemsetAsync(colsq,  0, HDIM * 4, stream);
        scatter_kernel<<<scatterBlocks, 256, 0, stream>>>(src, dst, order, h16, agg3, E, N);
        layer_fused_kernel<<<mblocks, 256, 0, stream>>>(
            agg3, packConv + (size_t)l * K1 * HDIM * HDIM, packMlp + (size_t)l * HDIM * HDIM,
            mlp_b + l * HDIM, vbuf, colsum, colsq, N, K1);
        bn_finalize_kernel<<<1, HDIM, 0, stream>>>(colsum, colsq, bn_g + l * HDIM, bn_b + l * HDIM, ss, N);
        bn_apply_kernel<<<(N * HDIM + 255) / 256, 256, 0, stream>>>(vbuf, ss, h16, N * HDIM);
    }

    // ---- pooling + classifier ----
    hipMemsetAsync(pooled, 0, (size_t)Bg * HDIM * 4, stream);
    pool_kernel<<<(int)(((size_t)N * 32 + 255) / 256), 256, 0, stream>>>(graphid, h16, pooled, N);
    classifier_kernel<<<1, 256, (size_t)Bg * Hh * 4, stream>>>(
        pooled, cls_W1, cls_b1, cls_W2, cls_b2, out, Bg, C, Hh);
}
